// WaveDensityAttentionBlock_17051020165576
// MI455X (gfx1250) — matrix operand order repaired and checked by reading, not compiled, not run on hardware
//
#include <hip/hip_runtime.h>
#include <math.h>

#define Bb   2
#define Ss   1024
#define Ee   1024
#define Hh   8
#define HDd  128
#define Mm   128
#define Ww   32
#define GHg  128
#define Rr   8
#define MRr  8
#define FFN  4096
#define ALPHA 3.0f
#define CMIX  0.25f

typedef __attribute__((ext_vector_type(16))) __bf16 v16bf;
typedef __attribute__((ext_vector_type(8)))  float  v8f;

union Frag { uint4 q[2]; v16bf v; unsigned short u[16]; };

__device__ __forceinline__ unsigned short f2bf(float f){
  unsigned int u = __float_as_uint(f);
  u += 0x7FFFu + ((u >> 16) & 1u);
  return (unsigned short)(u >> 16);
}
__device__ __forceinline__ float geluf(float v){
  return 0.5f * v * (1.0f + tanhf(0.7978845608f * (v + 0.044715f * v * v * v)));
}
// async global -> LDS 16B copy (CDNA5 TDM-adjacent path, ASYNCcnt tracked).
// ldsOff = low 32 bits of the flat shared address (LDS aperture keeps offset in addr[31:0]).
__device__ __forceinline__ void async_g2l_b128(unsigned ldsOff, const void* gptr){
  asm volatile("global_load_async_to_lds_b128 %0, %1, off"
               :: "v"(ldsOff), "v"((unsigned long long)(size_t)gptr) : "memory");
}
__device__ __forceinline__ void async_wait0(){
  asm volatile("s_wait_asynccnt 0x0" ::: "memory");
}

// ---------------- conversion kernels ----------------
__global__ void conv_bf(const float* __restrict__ in, unsigned short* __restrict__ out, int n){
  int i = blockIdx.x * blockDim.x + threadIdx.x;
  if (i < n) out[i] = f2bf(in[i]);
}
// in is (Kd x Nd) row-major f32; out is (Nd x Kd) row-major bf16
__global__ void transp_bf(const float* __restrict__ in, unsigned short* __restrict__ out, int Kd, int Nd){
  int i = blockIdx.x * blockDim.x + threadIdx.x;
  if (i >= Kd * Nd) return;
  int n = i / Kd, k = i % Kd;
  out[i] = f2bf(in[(size_t)k * Nd + n]);
}

// ---------------- pooled mean over S ----------------
__global__ void pooled_kernel(const float* __restrict__ x, float* __restrict__ pooled){
  int idx = blockIdx.x * blockDim.x + threadIdx.x; // b*E+e
  if (idx >= Bb * Ee) return;
  int b = idx >> 10, e = idx & 1023;
  float s = 0.f;
  for (int t = 0; t < Ss; ++t) s += x[((size_t)b * Ss + t) * Ee + e];
  pooled[idx] = s * (1.0f / Ss);
}

// ---------------- gate / mod -> wmod (B,H,M) ----------------
__global__ __launch_bounds__(128)
void gate_mod(const float* __restrict__ pooled,
              const float* __restrict__ gw1, const float* __restrict__ gb1,
              const float* __restrict__ gw2, const float* __restrict__ gb2,
              const float* __restrict__ mw1, const float* __restrict__ mb1,
              const float* __restrict__ mw2, const float* __restrict__ mb2,
              const float* __restrict__ mbasis, float* __restrict__ wmod){
  __shared__ float gh[GHg], mh[GHg], red[128], mr8[MRr];
  const int bh = blockIdx.x; const int b = bh >> 3, h = bh & 7;
  const int tid = threadIdx.x;
  const float* pb = pooled + (size_t)b * Ee;
  float a1 = 0.f, a2 = 0.f;
  for (int e = 0; e < Ee; ++e){
    float pv = pb[e];
    a1 += pv * gw1[((size_t)h * Ee + e) * GHg + tid];
    a2 += pv * mw1[((size_t)h * Ee + e) * GHg + tid];
  }
  gh[tid] = fmaxf(a1 + gb1[h * GHg + tid], 0.f);
  mh[tid] = fmaxf(a2 + mb1[h * GHg + tid], 0.f);
  __syncthreads();
  float gl = 0.f;
  for (int g = 0; g < GHg; ++g) gl += gh[g] * gw2[((size_t)h * GHg + g) * Mm + tid];
  gl += gb2[h * Mm + tid];
  if (tid < MRr){
    float mrv = 0.f;
    for (int g = 0; g < GHg; ++g) mrv += mh[g] * mw2[((size_t)h * GHg + g) * MRr + tid];
    mr8[tid] = mrv + mb2[h * MRr + tid];
  }
  __syncthreads();
  // softmax over the 128 gate logits (GATE_TEMP==1; straight-through == soft in fwd)
  red[tid] = gl; __syncthreads();
  for (int o = 64; o > 0; o >>= 1){ if (tid < o) red[tid] = fmaxf(red[tid], red[tid + o]); __syncthreads(); }
  float mx = red[0]; __syncthreads();
  float ex = __expf(gl - mx);
  red[tid] = ex; __syncthreads();
  for (int o = 64; o > 0; o >>= 1){ if (tid < o) red[tid] += red[tid + o]; __syncthreads(); }
  float soft = ex / red[0];
  float mo = 0.f;
  for (int r = 0; r < MRr; ++r) mo += mr8[r] * mbasis[((size_t)h * MRr + r) * Mm + tid];
  wmod[(size_t)bh * Mm + tid] = soft * (1.f + mo);
}

// ---------------- wave kernel table, reversed: karr[h,m,d] = kernels[h,m,S-1-d] ----------------
__global__ void wave_karr(const float* __restrict__ freqs, const float* __restrict__ amps,
                          const float* __restrict__ phases, float* __restrict__ karr){
  int idx = blockIdx.x * blockDim.x + threadIdx.x;
  if (idx >= Hh * Mm * Ss) return;
  int d = idx & (Ss - 1);
  int hm = idx >> 10;
  float p1 = -(float)d * (1.0f / Ss);                                     // rel = -d
  float p2 = (d == 0) ? 0.f : -sqrtf((float)d + 1e-6f) * (1.0f / 32.0f);  // sign(rel)*sqrt(|rel|+eps)/sqrt(S)
  float acc = 0.f;
  const float* fq = freqs  + (size_t)hm * Ww * 2;
  const float* am = amps   + (size_t)hm * Ww;
  const float* ph = phases + (size_t)hm * Ww;
  for (int w = 0; w < Ww; ++w){
    float dt = p1 * fq[2 * w] + p2 * fq[2 * w + 1];
    acc += sinf(6.2831853072f * dt + ph[w]) * am[w];
  }
  karr[idx] = acc;
}

// ---------------- ker2[b,h,d] = ALPHA * win[d] * sum_m wmod * karr ----------------
__global__ void ker2_kernel(const float* __restrict__ wmod, const float* __restrict__ karr,
                            float* __restrict__ ker2){
  int idx = blockIdx.x * blockDim.x + threadIdx.x;
  if (idx >= Bb * Hh * Ss) return;
  int d = idx & (Ss - 1);
  int bh = idx >> 10;
  int h = bh & 7;
  float win = __expf(-(float)d * (float)d * (1.0f / 131072.0f)); // 2*(S/4)^2
  float acc = 0.f;
  const float* wm = wmod + (size_t)bh * Mm;
  const float* kb = karr + (size_t)h * Mm * Ss + d;
  for (int m2 = 0; m2 < Mm; ++m2) acc += wm[m2] * kb[(size_t)m2 * Ss];
  ker2[idx] = ALPHA * win * acc;
}

// ---------------- per-row qs, km, qc, kc ----------------
__global__ __launch_bounds__(128)
void qskm_qckc(const float* __restrict__ x,
               const float* __restrict__ qw, const float* __restrict__ qb,
               const float* __restrict__ kw, const float* __restrict__ kb,
               const float* __restrict__ cwq, const float* __restrict__ cwk,
               float* __restrict__ qs, float* __restrict__ km,
               float* __restrict__ qc, float* __restrict__ kc){
  __shared__ float xrow[Ee];
  __shared__ float rq[128], rk[128];
  const int bs = blockIdx.x;            // b*S + s
  const int b = bs >> 10, s = bs & 1023;
  const int tid = threadIdx.x;
  for (int i = tid; i < Ee; i += 128) xrow[i] = x[(size_t)bs * Ee + i];
  __syncthreads();
  const int h = tid >> 4, seg = tid & 15;
  float pq = 0.f, pk = 0.f;
  for (int e = seg * 64; e < seg * 64 + 64; ++e){
    float xv = xrow[e];
    pq += xv * qw[(size_t)e * Hh + h];
    pk += xv * kw[(size_t)e * Hh + h];
  }
  rq[tid] = pq; rk[tid] = pk; __syncthreads();
  for (int o = 8; o > 0; o >>= 1){
    if (seg < o){ rq[tid] += rq[tid + o]; rk[tid] += rk[tid + o]; }
    __syncthreads();
  }
  if (seg == 0){
    qs[(size_t)bs * Hh + h] = rq[tid] + qb[h];
    km[(size_t)bs * Hh + h] = rk[tid] + kb[h];
  }
  if (seg < Rr){
    const int r = seg;
    float aq = 0.f, ak = 0.f;
    for (int hd = 0; hd < HDd; ++hd){
      float xv = xrow[h * HDd + hd];
      aq += xv * cwq[((size_t)h * HDd + hd) * Rr + r];
      ak += xv * cwk[((size_t)h * HDd + hd) * Rr + r];
    }
    size_t oi = (((size_t)(b * Hh + h)) * Ss + s) * Rr + r;
    qc[oi] = aq; kc[oi] = ak;
  }
}

// ---------------- tiled bf16 WMMA GEMM: C = A(MxK) * Bt(NxK)^T + bias ----------------
// Double-buffered LDS, async global->LDS staging (ASYNCcnt), wave32 WMMA fragments.
// MODE 0: store f32. MODE 1: store bf16 transposed to (B,H,HD,S) (v-proj). MODE 2: gelu, store bf16.
template<int MODE>
__global__ __launch_bounds__(256)
void gemm_bf16(const unsigned short* __restrict__ Ag,
               const unsigned short* __restrict__ Btg,
               const float* __restrict__ bias,
               float* __restrict__ outF,
               unsigned short* __restrict__ outBF,
               int N, int K){
  __shared__ __align__(16) unsigned short As[2][128 * 40];
  __shared__ __align__(16) unsigned short Bs[2][128 * 40];
  const int tid = threadIdx.x;
  const int lane = tid & 31, wid = tid >> 5;
  const int wm = wid >> 2, wn = wid & 3;
  const int hs = (lane >> 4) & 1, ln16 = lane & 15;
  const int m0 = blockIdx.y * 128;
  const int n0 = blockIdx.x * 128;

  // per-thread staging coordinates: 512 16B chunks per tile, 2 per thread
  const int r0 = tid >> 2,          c0 = (tid & 3) * 8;
  const int r1 = (tid + 256) >> 2,  c1 = ((tid + 256) & 3) * 8;

  v8f acc[4][2];
  for (int mi = 0; mi < 4; ++mi)
    for (int ni = 0; ni < 2; ++ni)
      acc[mi][ni] = (v8f){0.f,0.f,0.f,0.f,0.f,0.f,0.f,0.f};

  // prologue: async-stage tile 0
  {
    async_g2l_b128((unsigned)(size_t)&As[0][r0 * 40 + c0], Ag  + (size_t)(m0 + r0) * K + c0);
    async_g2l_b128((unsigned)(size_t)&Bs[0][r0 * 40 + c0], Btg + (size_t)(n0 + r0) * K + c0);
    async_g2l_b128((unsigned)(size_t)&As[0][r1 * 40 + c1], Ag  + (size_t)(m0 + r1) * K + c1);
    async_g2l_b128((unsigned)(size_t)&Bs[0][r1 * 40 + c1], Btg + (size_t)(n0 + r1) * K + c1);
  }
  async_wait0();
  __syncthreads();

  for (int k0 = 0; k0 < K; k0 += 32){
    const int buf = (k0 >> 5) & 1;
    if (k0 + 32 < K){ // prefetch next tile into the other buffer while computing this one
      const int nb = buf ^ 1; const int kn = k0 + 32;
      async_g2l_b128((unsigned)(size_t)&As[nb][r0 * 40 + c0], Ag  + (size_t)(m0 + r0) * K + kn + c0);
      async_g2l_b128((unsigned)(size_t)&Bs[nb][r0 * 40 + c0], Btg + (size_t)(n0 + r0) * K + kn + c0);
      async_g2l_b128((unsigned)(size_t)&As[nb][r1 * 40 + c1], Ag  + (size_t)(m0 + r1) * K + kn + c1);
      async_g2l_b128((unsigned)(size_t)&Bs[nb][r1 * 40 + c1], Btg + (size_t)(n0 + r1) * K + kn + c1);
    }
    Frag af[4], bfr[2];
    #pragma unroll
    for (int mi = 0; mi < 4; ++mi){
      const unsigned short* rp = &As[buf][(wm * 64 + mi * 16 + ln16) * 40];
      af[mi].q[0] = *(const uint4*)(rp + hs * 8);        // K = hs*8 .. +7
      af[mi].q[1] = *(const uint4*)(rp + 16 + hs * 8);   // K = 16+hs*8 .. +7
    }
    #pragma unroll
    for (int ni = 0; ni < 2; ++ni){
      const unsigned short* rp = &Bs[buf][(wn * 32 + ni * 16 + ln16) * 40];
      bfr[ni].q[0] = *(const uint4*)(rp + hs * 8);       // K = hs*8 .. +7
      bfr[ni].q[1] = *(const uint4*)(rp + 16 + hs * 8);  // K = 16+hs*8 .. +7
    }
    #pragma unroll
    for (int mi = 0; mi < 4; ++mi)
      #pragma unroll
      for (int ni = 0; ni < 2; ++ni)
        acc[mi][ni] = __builtin_amdgcn_wmma_f32_16x16x32_bf16(
            false, af[mi].v, false, bfr[ni].v, (short)0, acc[mi][ni], false, false);
    async_wait0();
    __syncthreads();
  }

  #pragma unroll
  for (int mi = 0; mi < 4; ++mi)
    #pragma unroll
    for (int ni = 0; ni < 2; ++ni){
      const int col = n0 + wn * 32 + ni * 16 + ln16;
      const float bv = bias ? bias[col] : 0.f;
      #pragma unroll
      for (int g = 0; g < 8; ++g){
        const int row = m0 + wm * 64 + mi * 16 + hs * 8 + g;
        float v = acc[mi][ni][g] + bv;
        if (MODE == 2) v = geluf(v);
        if (MODE == 0) outF[(size_t)row * N + col] = v;
        if (MODE == 2) outBF[(size_t)row * N + col] = f2bf(v);
        if (MODE == 1){
          int bb = row >> 10, ss = row & 1023;
          int hh = col >> 7,  hd = col & 127;
          outBF[(((size_t)(bb * Hh + hh)) * HDd + hd) * Ss + ss] = f2bf(v);
        }
      }
    }
}

// ---------------- fused flash attention (per 128-row tile of one (b,h)) ----------------
__global__ __launch_bounds__(256)
void attention_kernel(const unsigned short* __restrict__ vT,   // (B,H,HD,S) bf16
                      const float* __restrict__ ker2,          // (B,H,S)
                      const float* __restrict__ qs,            // (B,S,H)
                      const float* __restrict__ km,            // (B,S,H)
                      const float* __restrict__ qc,            // (B,H,S,R)
                      const float* __restrict__ kc,            // (B,H,S,R)
                      unsigned short* __restrict__ oBF){       // (B,S,E) bf16
  __shared__ __align__(16) unsigned short Vs[128 * 40];
  __shared__ float kms[32];
  __shared__ float kcs[32 * 8];
  const int tile = blockIdx.x, h = blockIdx.y, b = blockIdx.z;
  const int tid = threadIdx.x, lane = tid & 31, wid = tid >> 5;
  const int hs = (lane >> 4) & 1, ln16 = lane & 15;
  const int rowbase = tile * 128;
  const int sr = rowbase + wid * 16 + ln16;   // this lane's softmax row (A-fragment layout)

  const float* ker2p = ker2 + (size_t)(b * Hh + h) * Ss;
  const float qs_s = qs[((size_t)b * Ss + sr) * Hh + h];
  float qcr[Rr];
  #pragma unroll
  for (int r = 0; r < Rr; ++r) qcr[r] = qc[(((size_t)(b * Hh + h)) * Ss + sr) * Rr + r];

  v8f acc[8];
  #pragma unroll
  for (int ni = 0; ni < 8; ++ni) acc[ni] = (v8f){0.f,0.f,0.f,0.f,0.f,0.f,0.f,0.f};
  float m = -INFINITY, lsum = 0.f;

  const unsigned short* vbase = vT + (size_t)(b * Hh + h) * HDd * Ss;
  const int nch = (rowbase + 128) >> 5;
  const int vr0 = tid >> 2,         vc0 = (tid & 3) * 8;
  const int vr1 = (tid + 256) >> 2, vc1 = ((tid + 256) & 3) * 8;

  for (int c = 0; c < nch; ++c){
    const int tb = c << 5;
    // async-stage V tile (HD=128 rows x 32 t) into LDS; km/kc via regular stores
    async_g2l_b128((unsigned)(size_t)&Vs[vr0 * 40 + vc0], vbase + (size_t)vr0 * Ss + tb + vc0);
    async_g2l_b128((unsigned)(size_t)&Vs[vr1 * 40 + vc1], vbase + (size_t)vr1 * Ss + tb + vc1);
    if (tid < 32) kms[tid] = km[((size_t)b * Ss + tb + tid) * Hh + h];
    { int t = tid >> 3, r = tid & 7;
      kcs[tid] = kc[(((size_t)(b * Hh + h)) * Ss + tb + t) * Rr + r]; }
    async_wait0();
    __syncthreads();

    // logits for this lane's 16 A-fragment slots
    float pj[16]; float cmax = -INFINITY;
    #pragma unroll
    for (int j = 0; j < 16; ++j){
      const int kkj = (j < 8) ? (hs * 8 + j) : (16 + hs * 8 + (j - 8));
      const int t = tb + kkj;
      float val = -INFINITY;
      if (t <= sr){
        const float* kr = &kcs[kkj * 8];
        float dc = 0.f;
        #pragma unroll
        for (int r = 0; r < Rr; ++r) dc += qcr[r] * kr[r];
        val = ker2p[sr - t] + qs_s + kms[kkj] + CMIX * dc;
      }
      pj[j] = val; cmax = fmaxf(cmax, val);
    }
    cmax = fmaxf(cmax, __shfl_xor(cmax, 16, 32));
    const float nm  = fmaxf(m, cmax);
    const float nms = (nm == -INFINITY) ? 0.f : nm;
    const float scale = __expf(m - nms);     // 0 when m == -inf (acc is still zero then)
    float ssum = 0.f;
    #pragma unroll
    for (int j = 0; j < 16; ++j){ float p = __expf(pj[j] - nms); pj[j] = p; ssum += p; }
    ssum += __shfl_xor(ssum, 16, 32);
    lsum = lsum * scale + ssum;
    m = nm;
    // rescale accumulators: C-layout rows need row-wise scale broadcast
    float scg[8];
    #pragma unroll
    for (int g = 0; g < 8; ++g) scg[g] = __shfl(scale, hs * 8 + g, 32);
    #pragma unroll
    for (int ni = 0; ni < 8; ++ni)
      #pragma unroll
      for (int g = 0; g < 8; ++g) acc[ni][g] *= scg[g];
    // pack P as bf16 A fragment and accumulate P @ V
    Frag pf;
    #pragma unroll
    for (int j = 0; j < 16; ++j) pf.u[j] = f2bf(pj[j]);
    #pragma unroll
    for (int ni = 0; ni < 8; ++ni){
      Frag vf;
      const unsigned short* rp = &Vs[(ni * 16 + ln16) * 40];
      vf.q[0] = *(const uint4*)(rp + hs * 8);
      vf.q[1] = *(const uint4*)(rp + 16 + hs * 8);
      acc[ni] = __builtin_amdgcn_wmma_f32_16x16x32_bf16(
          false, pf.v, false, vf.v, (short)0, acc[ni], false, false);
    }
    __syncthreads();
  }

  // normalize and store o (bf16) for the out-projection GEMM
  #pragma unroll
  for (int g = 0; g < 8; ++g){
    const float lr = __shfl(lsum, hs * 8 + g, 32);
    const float inv = 1.f / lr;
    const int row = rowbase + wid * 16 + hs * 8 + g;
    #pragma unroll
    for (int ni = 0; ni < 8; ++ni){
      const int col = h * HDd + ni * 16 + ln16;
      oBF[((size_t)b * Ss + row) * Ee + col] = f2bf(acc[ni][g] * inv);
    }
  }
}

// ---------------- residual + layernorm ----------------
__global__ __launch_bounds__(256)
void resid_ln(const float* __restrict__ in1, const float* __restrict__ in2,
              const float* __restrict__ g, const float* __restrict__ bb,
              float* __restrict__ outF, unsigned short* __restrict__ outBF){
  __shared__ float zrow[Ee];
  __shared__ float r1[256], r2[256];
  const size_t row = blockIdx.x;
  const int tid = threadIdx.x;
  float s1 = 0.f, s2 = 0.f;
  for (int i = tid; i < Ee; i += 256){
    float z = in1[row * Ee + i] + in2[row * Ee + i];
    zrow[i] = z; s1 += z; s2 += z * z;
  }
  r1[tid] = s1; r2[tid] = s2; __syncthreads();
  for (int o = 128; o > 0; o >>= 1){
    if (tid < o){ r1[tid] += r1[tid + o]; r2[tid] += r2[tid + o]; }
    __syncthreads();
  }
  const float mean = r1[0] * (1.0f / Ee);
  const float var  = r2[0] * (1.0f / Ee) - mean * mean;
  const float rstd = rsqrtf(var + 1e-5f);
  for (int i = tid; i < Ee; i += 256){
    float val = (zrow[i] - mean) * rstd * g[i] + bb[i];
    outF[row * Ee + i] = val;
    if (outBF) outBF[row * Ee + i] = f2bf(val);
  }
}

// ---------------- host side ----------------
extern "C" void kernel_launch(void* const* d_in, const int* in_sizes, int n_in,
                              void* d_out, int out_size, void* d_ws, size_t ws_size,
                              hipStream_t stream){
  const float* x        = (const float*)d_in[0];
  const float* v_w      = (const float*)d_in[1];
  const float* v_b      = (const float*)d_in[2];
  const float* out_w    = (const float*)d_in[3];
  const float* out_b    = (const float*)d_in[4];
  const float* qmod_w   = (const float*)d_in[5];
  const float* qmod_b   = (const float*)d_in[6];
  const float* kmod_w   = (const float*)d_in[7];
  const float* kmod_b   = (const float*)d_in[8];
  const float* gate_w1  = (const float*)d_in[9];
  const float* gate_b1  = (const float*)d_in[10];
  const float* gate_w2  = (const float*)d_in[11];
  const float* gate_b2  = (const float*)d_in[12];
  const float* mod_w1   = (const float*)d_in[13];
  const float* mod_b1   = (const float*)d_in[14];
  const float* mod_w2   = (const float*)d_in[15];
  const float* mod_b2   = (const float*)d_in[16];
  const float* mod_basis= (const float*)d_in[17];
  const float* freqs    = (const float*)d_in[18];
  const float* amps     = (const float*)d_in[19];
  const float* phases   = (const float*)d_in[20];
  const float* cwq      = (const float*)d_in[21];
  const float* cwk      = (const float*)d_in[22];
  const float* ffn_w1   = (const float*)d_in[23];
  const float* ffn_b1   = (const float*)d_in[24];
  const float* ffn_w2   = (const float*)d_in[25];
  const float* ffn_b2   = (const float*)d_in[26];
  const float* ln_a_g   = (const float*)d_in[27];
  const float* ln_a_b   = (const float*)d_in[28];
  const float* ln_f_g   = (const float*)d_in[29];
  const float* ln_f_b   = (const float*)d_in[30];
  float* outp = (float*)d_out;

  char* ws = (char*)d_ws;
  size_t off = 0;
  auto alloc = [&](size_t bytes)->char*{
    char* p = ws + off;
    off += (bytes + 255) & ~(size_t)255;
    return p;
  };
  const size_t MROWS = (size_t)Bb * Ss;                 // 2048
  unsigned short* x_bf    = (unsigned short*)alloc(MROWS * Ee * 2);
  unsigned short* vwT     = (unsigned short*)alloc((size_t)Ee * Ee * 2);
  unsigned short* owT     = (unsigned short*)alloc((size_t)Ee * Ee * 2);
  unsigned short* f1T     = (unsigned short*)alloc((size_t)FFN * Ee * 2);
  unsigned short* f2T     = (unsigned short*)alloc((size_t)Ee * FFN * 2);
  float*          pooled  = (float*)alloc((size_t)Bb * Ee * 4);
  float*          wmod    = (float*)alloc((size_t)Bb * Hh * Mm * 4);
  float*          karr    = (float*)alloc((size_t)Hh * Mm * Ss * 4);
  float*          ker2b   = (float*)alloc((size_t)Bb * Hh * Ss * 4);
  float*          qsb     = (float*)alloc(MROWS * Hh * 4);
  float*          kmb     = (float*)alloc(MROWS * Hh * 4);
  float*          qcb     = (float*)alloc((size_t)Bb * Hh * Ss * Rr * 4);
  float*          kcb     = (float*)alloc((size_t)Bb * Hh * Ss * Rr * 4);
  unsigned short* vT_bf   = (unsigned short*)alloc((size_t)Bb * Hh * HDd * Ss * 2);
  unsigned short* o_bf    = (unsigned short*)alloc(MROWS * Ee * 2);
  float*          oproj   = (float*)alloc(MROWS * Ee * 4);
  float*          yF      = (float*)alloc(MROWS * Ee * 4);
  unsigned short* yBF     = (unsigned short*)alloc(MROWS * Ee * 2);
  unsigned short* hBF     = (unsigned short*)alloc(MROWS * FFN * 2);
  float*          fF      = (float*)alloc(MROWS * Ee * 4);
  (void)ws_size; (void)n_in; (void)in_sizes; (void)out_size;

  // 1. conversions
  { int n = (int)(MROWS * Ee);
    conv_bf<<<(n + 255) / 256, 256, 0, stream>>>(x, x_bf, n); }
  { int n = Ee * Ee;
    transp_bf<<<(n + 255) / 256, 256, 0, stream>>>(v_w,   vwT, Ee, Ee);
    transp_bf<<<(n + 255) / 256, 256, 0, stream>>>(out_w, owT, Ee, Ee); }
  { int n = Ee * FFN;
    transp_bf<<<(n + 255) / 256, 256, 0, stream>>>(ffn_w1, f1T, Ee, FFN);
    transp_bf<<<(n + 255) / 256, 256, 0, stream>>>(ffn_w2, f2T, FFN, Ee); }

  // 2. pooled + gate/mod
  pooled_kernel<<<(Bb * Ee + 255) / 256, 256, 0, stream>>>(x, pooled);
  gate_mod<<<Bb * Hh, 128, 0, stream>>>(pooled, gate_w1, gate_b1, gate_w2, gate_b2,
                                        mod_w1, mod_b1, mod_w2, mod_b2, mod_basis, wmod);

  // 3. wave kernel table + combined positional score table
  { int n = Hh * Mm * Ss;
    wave_karr<<<(n + 255) / 256, 256, 0, stream>>>(freqs, amps, phases, karr); }
  { int n = Bb * Hh * Ss;
    ker2_kernel<<<(n + 255) / 256, 256, 0, stream>>>(wmod, karr, ker2b); }

  // 4. per-row modulation & content projections
  qskm_qckc<<<(int)MROWS, 128, 0, stream>>>(x, qmod_w, qmod_b, kmod_w, kmod_b,
                                            cwq, cwk, qsb, kmb, qcb, kcb);

  // 5. v projection (stores V transposed per head as bf16 for attention)
  gemm_bf16<1><<<dim3(Ee / 128, (int)(MROWS / 128)), 256, 0, stream>>>(
      x_bf, vwT, v_b, nullptr, vT_bf, Ee, Ee);

  // 6. fused attention
  attention_kernel<<<dim3(Ss / 128, Hh, Bb), 256, 0, stream>>>(
      vT_bf, ker2b, qsb, kmb, qcb, kcb, o_bf);

  // 7. out projection
  gemm_bf16<0><<<dim3(Ee / 128, (int)(MROWS / 128)), 256, 0, stream>>>(
      o_bf, owT, out_b, oproj, nullptr, Ee, Ee);

  // 8. residual + LN (attn)
  resid_ln<<<(int)MROWS, 256, 0, stream>>>(x, oproj, ln_a_g, ln_a_b, yF, yBF);

  // 9. FFN
  gemm_bf16<2><<<dim3(FFN / 128, (int)(MROWS / 128)), 256, 0, stream>>>(
      yBF, f1T, ffn_b1, nullptr, hBF, FFN, Ee);
  gemm_bf16<0><<<dim3(Ee / 128, (int)(MROWS / 128)), 256, 0, stream>>>(
      hBF, f2T, ffn_b2, fF, nullptr, Ee, FFN);

  // 10. residual + LN (ffn) -> output
  resid_ln<<<(int)MROWS, 256, 0, stream>>>(yF, fF, ln_f_g, ln_f_b, outp, nullptr);
}
